// SharedLinearSelfAttention_56332791054918
// MI455X (gfx1250) — matrix operand order repaired and checked by reading, not compiled, not run on hardware
//
#include <hip/hip_runtime.h>

typedef __bf16 bf16;
typedef __attribute__((ext_vector_type(16))) __bf16 v16bf;
typedef __attribute__((ext_vector_type(8)))  __bf16 v8bf;
typedef __attribute__((ext_vector_type(4)))  __bf16 v4bf;
typedef __attribute__((ext_vector_type(8)))  float  v8f;
typedef unsigned int v4u __attribute__((ext_vector_type(4)));
typedef int          v4i __attribute__((ext_vector_type(4)));
typedef int          v8i __attribute__((ext_vector_type(8)));

#define D_MODEL 1024
#define N_ROWS  32768   /* B*T = 4*8192 */
#define N_HEADS 16
#define D_HEAD  64
#define RANK    16
#define EPSV    1e-5f

/* ---- GEMM tiling ---- */
#define TM 128
#define TN 128
#define TK 32
#define ASTR 40   /* TK + 8 pad elems = 80B rows; TDM pads 16B after each 64B row */

/* =============== LDS byte-offset of a __shared__ object =============== */
typedef __attribute__((address_space(3))) char lds_char_t;
__device__ __forceinline__ unsigned lds_off(void* p)
{
    return (unsigned)(unsigned long long)(lds_char_t*)p;
}

/* =============== Tensor Data Mover: 2-D tile -> LDS ===============
   dsz: 0=1B,1=2B,2=4B.  tile_x elems/row, tile_y rows, stride in elems.
   pad_en with pad_int/pad_amt codes per D# group1 (ISA 08 §8.4). */
__device__ __forceinline__ void tdm_load_2d(unsigned lds, const void* g,
                                            unsigned dsz,
                                            unsigned tile_x, unsigned tile_y,
                                            unsigned tdim0, unsigned tdim1,
                                            unsigned stride,
                                            unsigned pad_en, unsigned pad_int, unsigned pad_amt)
{
    unsigned long long ga = (unsigned long long)g;
    v4u g0;
    g0[0] = 1u;                                        /* count=1 (valid)        */
    g0[1] = lds;                                       /* lds_addr               */
    g0[2] = (unsigned)(ga & 0xffffffffu);              /* global_addr[31:0]      */
    g0[3] = (unsigned)((ga >> 32) & 0x01ffffffu)       /* global_addr[56:32]     */
          | 0x80000000u;                               /* type=2 ("image")       */
    v8i g1;
    g1[0] = (int)((dsz << 16) | (pad_en << 20) | (pad_int << 22) | (pad_amt << 25));
    g1[1] = (int)((tdim0 & 0xffffu) << 16);            /* tensor_dim0[15:0]      */
    g1[2] = (int)(((tdim0 >> 16) & 0xffffu) | ((tdim1 & 0xffffu) << 16));
    g1[3] = (int)(((tdim1 >> 16) & 0xffffu) | ((tile_x & 0xffffu) << 16));
    g1[4] = (int)(tile_y & 0xffffu);                   /* tile_dim1; tile_dim2=0 */
    g1[5] = (int)stride;                               /* tensor_dim0_stride lo  */
    g1[6] = 0;                                         /* stride hi / dim1_stride*/
    g1[7] = 0;
    v4i z4 = {0, 0, 0, 0};
#if __clang_major__ >= 23
    v8i z8 = {0, 0, 0, 0, 0, 0, 0, 0};
    __builtin_amdgcn_tensor_load_to_lds(g0, g1, z4, z4, z8, 0);
#else
    __builtin_amdgcn_tensor_load_to_lds(g0, g1, z4, z4, 0);
#endif
}

/* =============== fp32 -> bf16 hi/lo split =============== */
__global__ void split_bf16(const float* __restrict__ in,
                           bf16* __restrict__ hi, bf16* __restrict__ lo, int n4)
{
    int i = blockIdx.x * blockDim.x + threadIdx.x;
    if (i >= n4) return;
    float4 f = ((const float4*)in)[i];
    bf16 h0 = (bf16)f.x, h1 = (bf16)f.y, h2 = (bf16)f.z, h3 = (bf16)f.w;
    v4bf hv; hv[0] = h0; hv[1] = h1; hv[2] = h2; hv[3] = h3;
    v4bf lv;
    lv[0] = (bf16)(f.x - (float)h0);
    lv[1] = (bf16)(f.y - (float)h1);
    lv[2] = (bf16)(f.z - (float)h2);
    lv[3] = (bf16)(f.w - (float)h3);
    ((v4bf*)hi)[i] = hv;
    ((v4bf*)lo)[i] = lv;
}

__global__ void zero_f32(float* p, int n)
{
    int i = blockIdx.x * blockDim.x + threadIdx.x;
    if (i < n) p[i] = 0.0f;
}

/* =============== LoRA intermediate: t = X @ down^T (N_ROWS x RANK) =============== */
__global__ __launch_bounds__(128) void lora_t(const float* __restrict__ X,
                                              const float* __restrict__ DownBase,
                                              const int* __restrict__ insIdx,
                                              float* __restrict__ T)
{
    __shared__ float xs[8 * D_MODEL];  /* 32 KB */
    int n0 = blockIdx.x * 8;
    if (threadIdx.x < 32) {
        tdm_load_2d(lds_off(xs), X + (size_t)n0 * D_MODEL, 2,
                    8 * D_MODEL, 1, 8 * D_MODEL, 1, 8 * D_MODEL, 0, 0, 0);
        __builtin_amdgcn_s_wait_tensorcnt(0);
    }
    __syncthreads();
    int r  = threadIdx.x & 15;
    int nn = threadIdx.x >> 4;
    const float* down = DownBase + (size_t)insIdx[0] * RANK * D_MODEL + (size_t)r * D_MODEL;
    const float* xr = xs + nn * D_MODEL;
    float acc = 0.0f;
    for (int d = 0; d < D_MODEL; d += 4) {
        float4 a = *(const float4*)(xr + d);
        float4 b = *(const float4*)(down + d);
        acc += a.x * b.x + a.y * b.y + a.z * b.z + a.w * b.w;
    }
    T[(size_t)(n0 + nn) * RANK + r] = acc;
}

/* =============== WMMA fragment loads from LDS =============== */
/* A 16x32 bf16: lanes 0-15 M=lane, elems {K0..7, K16..23}; lanes 16-31 M=lane-16, {K8..15, K24..31} */
__device__ __forceinline__ v16bf frag_a(const bf16* base, int lane, int row0)
{
    int m  = row0 + (lane & 15);
    int hk = (lane >> 4) << 3;                 /* 0 or 8 */
    const bf16* p = base + m * ASTR + hk;
    v8bf x0 = *(const v8bf*)(p);
    v8bf x1 = *(const v8bf*)(p + 16);
    v16bf r;
#pragma unroll
    for (int i = 0; i < 8; ++i) { r[i] = x0[i]; r[i + 8] = x1[i]; }
    return r;
}
/* B 32x16 bf16: lanes 0-15 N=lane, elems {K0..7, K16..23}; lanes 16-31 N=lane-16, {K8..15, K24..31} */
__device__ __forceinline__ v16bf frag_b(const bf16* base, int lane)
{
    int c  = lane & 15;
    int hk = (lane >> 4) << 3;                 /* 0 or 8 */
    const bf16* p = base + c * ASTR + hk;
    v8bf x0 = *(const v8bf*)(p);
    v8bf x1 = *(const v8bf*)(p + 16);
    v16bf r;
#pragma unroll
    for (int i = 0; i < 8; ++i) { r[i] = x0[i]; r[i + 8] = x1[i]; }
    return r;
}

/* =============== GEMM: Out = A @ W^T + t @ up^T, bf16x3 split-precision ===============
   mode 0: gelu1p   mode 1: gelu1p * mask   mode 2: * mask   mode 3: none */
__global__ __launch_bounds__(256) void gemm_bf16x3(
    const bf16* __restrict__ Ahi, const bf16* __restrict__ Alo,
    const bf16* __restrict__ Whi, const bf16* __restrict__ Wlo,
    const float* __restrict__ Trk, const float* __restrict__ UpBase,
    const int* __restrict__ insIdx, const unsigned char* __restrict__ mask,
    float* __restrict__ Out, int mode)
{
    __shared__ bf16 As_hi[TM * ASTR];
    __shared__ bf16 As_lo[TM * ASTR];
    __shared__ bf16 Bs_hi[TN * ASTR];
    __shared__ bf16 Bs_lo[TN * ASTR];
    __shared__ float Ts[TM * RANK];

    const int tid  = threadIdx.x;
    const int m0   = blockIdx.x * TN;
    const int n0   = blockIdx.y * TM;
    const int wv   = tid >> 5;                 /* wave32: 8 waves, one 16-col strip each */
    const int lane = tid & 31;
    const int hi   = lane >> 4;

    v8f acc[8] = {};

    for (int kb = 0; kb < D_MODEL; kb += TK) {
        if (tid < 32) {
            /* TDM: four 128x32 bf16 tiles, 64B rows padded +16B in LDS (ASTR=40) */
            tdm_load_2d(lds_off(As_hi), Ahi + (size_t)n0 * D_MODEL + kb, 1,
                        TK, TM, D_MODEL, N_ROWS, D_MODEL, 1, 3, 3);
            tdm_load_2d(lds_off(As_lo), Alo + (size_t)n0 * D_MODEL + kb, 1,
                        TK, TM, D_MODEL, N_ROWS, D_MODEL, 1, 3, 3);
            tdm_load_2d(lds_off(Bs_hi), Whi + (size_t)m0 * D_MODEL + kb, 1,
                        TK, TN, D_MODEL, D_MODEL, D_MODEL, 1, 3, 3);
            tdm_load_2d(lds_off(Bs_lo), Wlo + (size_t)m0 * D_MODEL + kb, 1,
                        TK, TN, D_MODEL, D_MODEL, D_MODEL, 1, 3, 3);
            __builtin_amdgcn_s_wait_tensorcnt(0);
        }
        __syncthreads();

        v16bf bh = frag_b(Bs_hi + wv * 16 * ASTR, lane);
        v16bf bl = frag_b(Bs_lo + wv * 16 * ASTR, lane);
#pragma unroll
        for (int r = 0; r < 8; ++r) {
            v16bf ah = frag_a(As_hi, lane, r * 16);
            v16bf al = frag_a(As_lo, lane, r * 16);
            acc[r] = __builtin_amdgcn_wmma_f32_16x16x32_bf16(false, ah, false, bh, (short)0, acc[r], false, false);
            acc[r] = __builtin_amdgcn_wmma_f32_16x16x32_bf16(false, ah, false, bl, (short)0, acc[r], false, false);
            acc[r] = __builtin_amdgcn_wmma_f32_16x16x32_bf16(false, al, false, bh, (short)0, acc[r], false, false);
        }
        __syncthreads();
    }

    /* stage LoRA t rows for this tile (contiguous 2048 f32) via TDM */
    if (tid < 32) {
        tdm_load_2d(lds_off(Ts), Trk + (size_t)n0 * RANK, 2,
                    TM * RANK, 1, TM * RANK, 1, TM * RANK, 0, 0, 0);
        __builtin_amdgcn_s_wait_tensorcnt(0);
    }
    __syncthreads();

    const int m = m0 + wv * 16 + (lane & 15);
    const float* up = UpBase + (size_t)insIdx[0] * D_MODEL * RANK + (size_t)m * RANK;
    float4 u0 = ((const float4*)up)[0];
    float4 u1 = ((const float4*)up)[1];
    float4 u2 = ((const float4*)up)[2];
    float4 u3 = ((const float4*)up)[3];

#pragma unroll
    for (int r = 0; r < 8; ++r) {
#pragma unroll
        for (int j = 0; j < 8; ++j) {
            int lr = r * 16 + hi * 8 + j;      /* local output row */
            int n  = n0 + lr;
            const float* tp = Ts + lr * RANK;
            float val = acc[r][j];
            val += tp[0]  * u0.x + tp[1]  * u0.y + tp[2]  * u0.z + tp[3]  * u0.w
                 + tp[4]  * u1.x + tp[5]  * u1.y + tp[6]  * u1.z + tp[7]  * u1.w
                 + tp[8]  * u2.x + tp[9]  * u2.y + tp[10] * u2.z + tp[11] * u2.w
                 + tp[12] * u3.x + tp[13] * u3.y + tp[14] * u3.z + tp[15] * u3.w;
            if (mode == 0) {
                val = 1.0f + 0.5f * val * (1.0f + erff(val * 0.70710678118654752f));
            } else if (mode == 1) {
                val = 1.0f + 0.5f * val * (1.0f + erff(val * 0.70710678118654752f));
                val *= mask[n] ? 1.0f : 0.0f;
            } else if (mode == 2) {
                val *= mask[n] ? 1.0f : 0.0f;
            }
            Out[(size_t)n * D_MODEL + m] = val;
        }
    }
}

/* =============== linear attention pass 1: ctx += k^T v, z += sum k =============== */
__global__ __launch_bounds__(256) void attn_p1(const float* __restrict__ Kb,
                                               const float* __restrict__ Vb,
                                               float* __restrict__ Ctx, float* __restrict__ Zb)
{
    __shared__ float ks[64 * D_HEAD];  /* 16 KB */
    __shared__ float vs[64 * D_HEAD];  /* 16 KB */
    int bh = blockIdx.y;
    int b = bh >> 4, h = bh & 15;
    int t0 = blockIdx.x * 64;
    size_t rowbase = (size_t)(b * 8192 + t0) * D_MODEL + h * D_HEAD;
    if (threadIdx.x < 32) {
        tdm_load_2d(lds_off(ks), Kb + rowbase, 2, D_HEAD, 64, D_MODEL, N_ROWS, D_MODEL, 0, 0, 0);
        tdm_load_2d(lds_off(vs), Vb + rowbase, 2, D_HEAD, 64, D_MODEL, N_ROWS, D_MODEL, 0, 0, 0);
        __builtin_amdgcn_s_wait_tensorcnt(0);
    }
    __syncthreads();
    int d0 = (threadIdx.x >> 4) << 2;
    int e0 = (threadIdx.x & 15) << 2;
    float acc[4][4] = {};
    for (int t = 0; t < 64; ++t) {
        const float* kp = ks + t * D_HEAD + d0;
        const float* vp = vs + t * D_HEAD + e0;
#pragma unroll
        for (int i = 0; i < 4; ++i)
#pragma unroll
            for (int j = 0; j < 4; ++j)
                acc[i][j] += kp[i] * vp[j];
    }
    float* ctx = Ctx + (size_t)bh * D_HEAD * D_HEAD;
#pragma unroll
    for (int i = 0; i < 4; ++i)
#pragma unroll
        for (int j = 0; j < 4; ++j)
            atomicAdd(&ctx[(d0 + i) * D_HEAD + e0 + j], acc[i][j]);
    if (threadIdx.x < D_HEAD) {
        float z = 0.0f;
        for (int t = 0; t < 64; ++t) z += ks[t * D_HEAD + threadIdx.x];
        atomicAdd(&Zb[bh * D_HEAD + threadIdx.x], z);
    }
}

/* =============== linear attention pass 2: out = (q @ ctx) / (q.z + eps) =============== */
__global__ __launch_bounds__(256) void attn_p2(const float* Qb,
                                               const float* __restrict__ Ctx,
                                               const float* __restrict__ Zb, float* Ob)
{
    __shared__ float cs[D_HEAD * D_HEAD];  /* 16 KB */
    __shared__ float qs[128 * D_HEAD];     /* 32 KB */
    __shared__ float zs[D_HEAD];
    __shared__ float den[128];
    int bh = blockIdx.y;
    int b = bh >> 4, h = bh & 15;
    int t0 = blockIdx.x * 128;
    size_t rowbase = (size_t)(b * 8192 + t0) * D_MODEL + h * D_HEAD;
    if (threadIdx.x < 32) {
        tdm_load_2d(lds_off(cs), Ctx + (size_t)bh * D_HEAD * D_HEAD, 2,
                    D_HEAD * D_HEAD, 1, D_HEAD * D_HEAD, 1, D_HEAD * D_HEAD, 0, 0, 0);
        tdm_load_2d(lds_off(qs), Qb + rowbase, 2, D_HEAD, 128, D_MODEL, N_ROWS, D_MODEL, 0, 0, 0);
        __builtin_amdgcn_s_wait_tensorcnt(0);
    }
    if (threadIdx.x < D_HEAD) zs[threadIdx.x] = Zb[bh * D_HEAD + threadIdx.x];
    __syncthreads();
    if (threadIdx.x < 128) {
        float d = 0.0f;
        const float* q = qs + threadIdx.x * D_HEAD;
        for (int e = 0; e < D_HEAD; ++e) d += q[e] * zs[e];
        den[threadIdx.x] = d;
    }
    __syncthreads();
    int t  = threadIdx.x >> 1;
    int c0 = (threadIdx.x & 1) << 5;
    float acc[32];
#pragma unroll
    for (int j = 0; j < 32; ++j) acc[j] = 0.0f;
    const float* q = qs + t * D_HEAD;
    for (int e = 0; e < D_HEAD; ++e) {
        float qv = q[e];
        const float* cr = cs + e * D_HEAD + c0;
#pragma unroll
        for (int j = 0; j < 32; ++j) acc[j] += qv * cr[j];
    }
    float invd = 1.0f / (den[t] + EPSV);
    float* o = Ob + rowbase + (size_t)t * D_MODEL + c0;
#pragma unroll
    for (int j = 0; j < 32; ++j) o[j] = acc[j] * invd;
}

/* =============== launch =============== */
extern "C" void kernel_launch(void* const* d_in, const int* in_sizes, int n_in,
                              void* d_out, int out_size, void* d_ws, size_t ws_size,
                              hipStream_t stream)
{
    (void)in_sizes; (void)n_in; (void)out_size; (void)ws_size;
    const float* x   = (const float*)d_in[0];
    const unsigned char* pad = (const unsigned char*)d_in[1];
    const int* ins   = (const int*)d_in[2];
    const float* w[4] = { (const float*)d_in[3], (const float*)d_in[4],
                          (const float*)d_in[5], (const float*)d_in[6] };
    const float* lqd = (const float*)d_in[7];
    const float* lqu = (const float*)d_in[8];
    const float* lkd = (const float*)d_in[9];
    const float* lku = (const float*)d_in[10];
    const float* lvd = (const float*)d_in[11];
    const float* lvu = (const float*)d_in[12];
    const float* lod = (const float*)d_in[13];
    const float* lou = (const float*)d_in[14];

    char* ws = (char*)d_ws;
    size_t off = 0;
    auto carve = [&](size_t bytes) -> void* {
        void* p = ws + off;
        off += (bytes + 255) & ~(size_t)255;
        return p;
    };
    const size_t XE = (size_t)N_ROWS * D_MODEL;      /* 33,554,432 elems */
    const size_t WE = (size_t)D_MODEL * D_MODEL;     /* 1,048,576 elems  */
    bf16*  x_hi = (bf16*)carve(XE * 2);
    bf16*  x_lo = (bf16*)carve(XE * 2);
    bf16*  w_hi = (bf16*)carve(4 * WE * 2);
    bf16*  w_lo = (bf16*)carve(4 * WE * 2);
    float* tbuf = (float*)carve((size_t)4 * N_ROWS * RANK * 4);
    float* qb   = (float*)carve(XE * 4);             /* q, later reused as attn output */
    float* kb   = (float*)carve(XE * 4);
    float* vb   = (float*)carve(XE * 4);
    float* ctx  = (float*)carve((size_t)64 * D_HEAD * D_HEAD * 4);
    float* zb   = (float*)carve((size_t)64 * D_HEAD * 4);

    const int nthr = 256;
    /* 1. split x and weights into bf16 hi/lo */
    split_bf16<<<(int)(XE / 4 + nthr - 1) / nthr, nthr, 0, stream>>>(x, x_hi, x_lo, (int)(XE / 4));
    for (int p = 0; p < 4; ++p)
        split_bf16<<<(int)(WE / 4 + nthr - 1) / nthr, nthr, 0, stream>>>(
            w[p], w_hi + p * WE, w_lo + p * WE, (int)(WE / 4));

    /* 2. LoRA intermediates for q,k,v */
    lora_t<<<N_ROWS / 8, 128, 0, stream>>>(x, lqd, ins, tbuf + 0 * (size_t)N_ROWS * RANK);
    lora_t<<<N_ROWS / 8, 128, 0, stream>>>(x, lkd, ins, tbuf + 1 * (size_t)N_ROWS * RANK);
    lora_t<<<N_ROWS / 8, 128, 0, stream>>>(x, lvd, ins, tbuf + 2 * (size_t)N_ROWS * RANK);

    /* 3. q/k/v projections (WMMA + TDM staging) with fused gelu1p / mask */
    dim3 gg(D_MODEL / TN, N_ROWS / TM, 1);
    gemm_bf16x3<<<gg, nthr, 0, stream>>>(x_hi, x_lo, w_hi + 0 * WE, w_lo + 0 * WE,
        tbuf + 0 * (size_t)N_ROWS * RANK, lqu, ins, pad, qb, 0);
    gemm_bf16x3<<<gg, nthr, 0, stream>>>(x_hi, x_lo, w_hi + 1 * WE, w_lo + 1 * WE,
        tbuf + 1 * (size_t)N_ROWS * RANK, lku, ins, pad, kb, 1);
    gemm_bf16x3<<<gg, nthr, 0, stream>>>(x_hi, x_lo, w_hi + 2 * WE, w_lo + 2 * WE,
        tbuf + 2 * (size_t)N_ROWS * RANK, lvu, ins, pad, vb, 2);

    /* 4. linear attention */
    int nz = 64 * D_HEAD * D_HEAD + 64 * D_HEAD;   /* ctx and zb are contiguous */
    zero_f32<<<(nz + nthr - 1) / nthr, nthr, 0, stream>>>(ctx, nz);
    attn_p1<<<dim3(128, 64, 1), nthr, 0, stream>>>(kb, vb, ctx, zb);
    attn_p2<<<dim3(64, 64, 1), nthr, 0, stream>>>(qb, ctx, zb, qb);  /* in-place into qb */

    /* 5. output projection */
    split_bf16<<<(int)(XE / 4 + nthr - 1) / nthr, nthr, 0, stream>>>(qb, x_hi, x_lo, (int)(XE / 4));
    lora_t<<<N_ROWS / 8, 128, 0, stream>>>(qb, lod, ins, tbuf + 3 * (size_t)N_ROWS * RANK);
    gemm_bf16x3<<<gg, nthr, 0, stream>>>(x_hi, x_lo, w_hi + 3 * WE, w_lo + 3 * WE,
        tbuf + 3 * (size_t)N_ROWS * RANK, lou, ins, nullptr, (float*)d_out, 3);
}
